// TimeAttention_56418690400781
// MI455X (gfx1250) — matrix operand order repaired and checked by reading, not compiled, not run on hardware
//
#include <hip/hip_runtime.h>
#include <hip/hip_bf16.h>
#include <hip/hip_fp16.h>

#define BB 4
#define SS 2048
#define DDIM 1024
#define HH 16
#define HD 64

typedef __attribute__((ext_vector_type(16))) _Float16 v16h;
typedef __attribute__((ext_vector_type(8)))  _Float16 v8h;
typedef __attribute__((ext_vector_type(8)))  float    v8f;
typedef __attribute__((ext_vector_type(4)))  int      v4i;

// ---------------------------------------------------------------------------
// Feature probes (fall back to VGPR-bounce copies if builtins are absent)
// ---------------------------------------------------------------------------
#if defined(__has_builtin)
#  if __has_builtin(__builtin_amdgcn_global_load_async_to_lds_b128)
#    define HAVE_ASYNC_COPY 1
#  else
#    define HAVE_ASYNC_COPY 0
#  endif
#  if __has_builtin(__builtin_amdgcn_s_wait_asynccnt)
#    define HAVE_WAIT_ASYNC 1
#  else
#    define HAVE_WAIT_ASYNC 0
#  endif
#else
#  define HAVE_ASYNC_COPY 0
#  define HAVE_WAIT_ASYNC 0
#endif

__device__ __forceinline__ void wait_async0() {
#if HAVE_WAIT_ASYNC
  __builtin_amdgcn_s_wait_asynccnt(0);
#else
  asm volatile("s_wait_asynccnt 0" ::: "memory");
#endif
}

// 16-byte global -> LDS copy (async path if available).
// Builtin signature (from hipcc diagnostic): param1 = v4i addrspace(1)*,
// so: (global v4i* src, lds v4i* dst, imm offset, imm cpol).
__device__ __forceinline__ void stage16(const _Float16* src, _Float16* dstLds) {
#if HAVE_ASYNC_COPY
  __builtin_amdgcn_global_load_async_to_lds_b128(
      (__attribute__((address_space(1))) v4i*)(v4i*)src,
      (__attribute__((address_space(3))) v4i*)(v4i*)dstLds, 0, 0);
#else
  *(v8h*)dstLds = *(const v8h*)src;
#endif
}

// ---------------------------------------------------------------------------
// WMMA helpers (CDNA5 16x16x32 f16 -> f32)
// ---------------------------------------------------------------------------
__device__ __forceinline__ v8f wmma_f16(v16h a, v16h b, v8f c) {
  return __builtin_amdgcn_wmma_f32_16x16x32_f16(
      /*neg_a=*/false, a, /*neg_b=*/false, b,
      /*c_mod=*/(short)0, c, /*reuse_a=*/false, /*reuse_b=*/false);
}

// A fragment: 16x32 (MxK) f16. Lane holds row m = lane&15,
// K = {kb..kb+7, 16+kb..16+kb+7} with kb = 8*(lane>>4). Two 16B loads.
__device__ __forceinline__ v16h load_frag_a(const _Float16* base, int ld, int lane) {
  const int m  = lane & 15;
  const int kb = (lane >> 4) << 3;
  const _Float16* p = base + (size_t)m * ld + kb;
  v8h lo = *(const v8h*)(p);
  v8h hi = *(const v8h*)(p + 16);
  v16h r;
#pragma unroll
  for (int i = 0; i < 8; ++i) { r[i] = lo[i]; r[i + 8] = hi[i]; }
  return r;
}

// B fragment: 32x16 (KxN) f16, B[k][n] = mem[n*ld + k]. Lane holds column
// n = lane&15, K = {kb..kb+7, 16+kb..16+kb+7} with kb = 8*(lane>>4). Two 16B loads.
__device__ __forceinline__ v16h load_frag_b(const _Float16* base, int ld, int lane) {
  const int n  = lane & 15;
  const int kb = (lane >> 4) << 3;
  const _Float16* p = base + (size_t)n * ld + kb;
  v8h lo = *(const v8h*)(p);
  v8h hi = *(const v8h*)(p + 16);
  v16h r;
#pragma unroll
  for (int i = 0; i < 8; ++i) { r[i] = lo[i]; r[i + 8] = hi[i]; }
  return r;
}

// ---------------------------------------------------------------------------
// fp32 -> fp16 conversion (grid-stride)
// ---------------------------------------------------------------------------
__global__ void cvt_f32_f16(const float* __restrict__ src, _Float16* __restrict__ dst, int n) {
  int i = blockIdx.x * blockDim.x + threadIdx.x;
  const int stride = gridDim.x * blockDim.x;
  for (; i < n; i += stride) dst[i] = (_Float16)src[i];
}

// ---------------------------------------------------------------------------
// GEMM: Y[M,N] = X[M,K] @ W^T + bias, W row-major [N,K].
// Block = 8 waves = 128x64 tile; the 64x32 B tile (from W) is staged in LDS
// once per k-step (async global->LDS, double buffered) and shared by all 8
// waves. Each wave computes a 16x64 strip (4 WMMA column tiles).
// OMODE 0: f16 row-major out; 1: f16 per-head-transposed out ([B,H,HD,S]);
// OMODE 2: f32 row-major out.
// ---------------------------------------------------------------------------
template <int OMODE>
__global__ void __launch_bounds__(256)
gemm_wmma(const _Float16* __restrict__ X, const _Float16* __restrict__ W,
          const float* __restrict__ bias, _Float16* __restrict__ Yh,
          float* __restrict__ Yf) {
  const int N = DDIM, K = DDIM;
  __shared__ _Float16 Bt[2][64 * 40];   // 64 cols x 32 k, row stride 40 f16

  const int lane = threadIdx.x & 31;
  const int wid  = threadIdx.x >> 5;
  const int br   = blockIdx.x >> 4;     // 64 row blocks of 128
  const int bc   = blockIdx.x & 15;     // 16 col blocks of 64
  const int row0 = br * 128 + wid * 16;
  const int col0 = bc * 64;

  // staging: thread t copies 16B; n = t>>2 (0..63), part = t&3
  const int sn = threadIdx.x >> 2;
  const int sp = (threadIdx.x & 3) * 8;
  const _Float16* wsrc = W + (size_t)(col0 + sn) * K + sp;
  _Float16* bdst0 = &Bt[0][sn * 40 + sp];
  _Float16* bdst1 = &Bt[1][sn * 40 + sp];

  stage16(wsrc, bdst0);                 // k-step 0 -> buffer 0

  v8f acc[4] = {};
  const int KSTEPS = K / 32;            // 32
  for (int ks = 0; ks < KSTEPS; ++ks) {
    wait_async0();
    __syncthreads();                    // staged tile visible; prev reads done
    if (ks + 1 < KSTEPS)
      stage16(wsrc + (size_t)(ks + 1) * 32, (ks & 1) ? bdst0 : bdst1);

    const _Float16* bbuf = (ks & 1) ? &Bt[1][0] : &Bt[0][0];
    v16h a = load_frag_a(X + (size_t)row0 * K + ks * 32, K, lane);
#pragma unroll
    for (int t = 0; t < 4; ++t) {
      v16h b = load_frag_b(bbuf + (size_t)(t * 16) * 40, 40, lane);
      acc[t] = wmma_f16(a, b, acc[t]);
    }
  }

  const int half = lane >> 4;
  const int nloc = lane & 15;
#pragma unroll
  for (int t = 0; t < 4; ++t) {
    const int col = col0 + t * 16 + nloc;
    const float bv = bias[col];
#pragma unroll
    for (int r = 0; r < 8; ++r) {
      const int row = row0 + r + 8 * half;   // C layout: VGPR r -> row r+8*half
      const float v = acc[t][r] + bv;
      if (OMODE == 0) {
        Yh[(size_t)row * N + col] = (_Float16)v;
      } else if (OMODE == 1) {
        const int b_ = row >> 11, s_ = row & (SS - 1);
        const int h_ = col >> 6,  d_ = col & (HD - 1);
        Yh[((size_t)(b_ * HH + h_) * HD + d_) * SS + s_] = (_Float16)v;
      } else {
        Yf[(size_t)row * N + col] = v;
      }
    }
  }
}

// ---------------------------------------------------------------------------
// Fused flash attention: one wave per (b, h, 16-query tile).
// 64 keys per iteration: 8 score WMMAs, online softmax (max butterfly only;
// the denominator is accumulated with a P @ ones WMMA), 8 P@V WMMAs.
// Q,K in [B*S, D] f16, V transposed as [B,H,HD,S] f16.
// ---------------------------------------------------------------------------
__global__ void __launch_bounds__(256)
flash_attn(const _Float16* __restrict__ Q, const _Float16* __restrict__ Kp,
           const _Float16* __restrict__ Vt, _Float16* __restrict__ Attn) {
  __shared__ _Float16 Plds[8][16 * 72];   // per-wave 16x64 P tile, padded rows
  const int lane = threadIdx.x & 31;
  const int widb = threadIdx.x >> 5;
  const int gwid = blockIdx.x * (blockDim.x >> 5) + widb;
  const int qt = gwid & 127;              // S/16 = 128 query tiles
  const int bh = gwid >> 7;               // 0..63
  const int b_ = bh >> 4, h_ = bh & 15;
  const int q0 = qt * 16;
  const float sc = 0.125f * 1.44269504088896340736f;  // 1/sqrt(64) * log2(e)

  const _Float16* Qb = Q  + ((size_t)(b_ * SS + q0)) * DDIM + h_ * HD;
  const _Float16* Kb = Kp + ((size_t)(b_ * SS)) * DDIM + h_ * HD;
  const _Float16* Vb = Vt + (size_t)bh * HD * SS;

  // Q fragments with softmax scale (and log2e) folded in
  v16h qf0 = load_frag_a(Qb,      DDIM, lane);
  v16h qf1 = load_frag_a(Qb + 32, DDIM, lane);
#pragma unroll
  for (int i = 0; i < 16; ++i) {
    qf0[i] = (_Float16)((float)qf0[i] * sc);
    qf1[i] = (_Float16)((float)qf1[i] * sc);
  }

  v16h onesf;
#pragma unroll
  for (int i = 0; i < 16; ++i) onesf[i] = (_Float16)1.0f;

  v8f acc[4] = {};
  v8f lfrag  = {};
  float mrun[8];
#pragma unroll
  for (int r = 0; r < 8; ++r) mrun[r] = -1e30f;

  _Float16* Pl = &Plds[widb][0];
  const int half = lane >> 4;
  const int nloc = lane & 15;

  for (int kk = 0; kk < SS; kk += 64) {
    // ---- scores: four 16x16 fragments over 64 keys, K-loop over HD=64 ----
    v8f c[4] = {};
#pragma unroll
    for (int j = 0; j < 4; ++j) {
      v16h b0 = load_frag_b(Kb + (size_t)(kk + 16 * j) * DDIM,      DDIM, lane);
      c[j] = wmma_f16(qf0, b0, c[j]);
      v16h b1 = load_frag_b(Kb + (size_t)(kk + 16 * j) * DDIM + 32, DDIM, lane);
      c[j] = wmma_f16(qf1, b1, c[j]);
    }
    // ---- running max (only reduction butterfly left) ----
    float mnew[8], alpha[8];
#pragma unroll
    for (int r = 0; r < 8; ++r) {
      float t = fmaxf(fmaxf(c[0][r], c[1][r]), fmaxf(c[2][r], c[3][r]));
#pragma unroll
      for (int msk = 1; msk < 16; msk <<= 1) t = fmaxf(t, __shfl_xor(t, msk, 32));
      mnew[r]  = fmaxf(mrun[r], t);
      alpha[r] = exp2f(mrun[r] - mnew[r]);
      mrun[r]  = mnew[r];
    }
    // ---- P = exp2(s - m) -> LDS (C layout -> row-major) ----
#pragma unroll
    for (int j = 0; j < 4; ++j) {
#pragma unroll
      for (int r = 0; r < 8; ++r) {
        const float p = exp2f(c[j][r] - mnew[r]);
        Pl[(r + 8 * half) * 72 + j * 16 + nloc] = (_Float16)p;
      }
    }
    asm volatile("s_wait_dscnt 0" ::: "memory");  // wave-private LDS RAW
    v16h pa0 = load_frag_a(Pl,      72, lane);
    v16h pa1 = load_frag_a(Pl + 32, 72, lane);

    // ---- denominator: l = l*alpha + P @ ones (two WMMAs, no shuffles) ----
#pragma unroll
    for (int r = 0; r < 8; ++r) lfrag[r] *= alpha[r];
    lfrag = wmma_f16(pa0, onesf, lfrag);
    lfrag = wmma_f16(pa1, onesf, lfrag);

    // ---- O = O*alpha + P @ V ----
#pragma unroll
    for (int t = 0; t < 4; ++t) {
#pragma unroll
      for (int r = 0; r < 8; ++r) acc[t][r] *= alpha[r];
      v16h bv0 = load_frag_b(Vb + (size_t)(t * 16) * SS + kk,      SS, lane);
      acc[t] = wmma_f16(pa0, bv0, acc[t]);
      v16h bv1 = load_frag_b(Vb + (size_t)(t * 16) * SS + kk + 32, SS, lane);
      acc[t] = wmma_f16(pa1, bv1, acc[t]);
    }
  }

  // ---- epilogue: normalize and store context (f16, [B*S, D]) ----
  float rl[8];
#pragma unroll
  for (int r = 0; r < 8; ++r) rl[r] = 1.0f / lfrag[r];
#pragma unroll
  for (int t = 0; t < 4; ++t) {
#pragma unroll
    for (int r = 0; r < 8; ++r) {
      const int row = b_ * SS + q0 + r + 8 * half;
      const int col = h_ * HD + t * 16 + nloc;
      Attn[(size_t)row * DDIM + col] = (_Float16)(acc[t][r] * rl[r]);
    }
  }
}

// ---------------------------------------------------------------------------
// Host launch
// ---------------------------------------------------------------------------
extern "C" void kernel_launch(void* const* d_in, const int* in_sizes, int n_in,
                              void* d_out, int out_size, void* d_ws, size_t ws_size,
                              hipStream_t stream) {
  const float* q  = (const float*)d_in[0];
  const float* k  = (const float*)d_in[1];
  const float* v  = (const float*)d_in[2];
  const float* Wq = (const float*)d_in[3];
  const float* bq = (const float*)d_in[4];
  const float* Wk = (const float*)d_in[5];
  const float* bk = (const float*)d_in[6];
  const float* Wv = (const float*)d_in[7];
  const float* bv = (const float*)d_in[8];
  const float* Wo = (const float*)d_in[9];
  const float* bo = (const float*)d_in[10];

  const size_t n_x = (size_t)BB * SS * DDIM;   // 8,388,608
  const size_t n_w = (size_t)DDIM * DDIM;      // 1,048,576

  _Float16* ws  = (_Float16*)d_ws;
  _Float16* Xq  = ws;
  _Float16* Xk  = Xq  + n_x;
  _Float16* Xv  = Xk  + n_x;
  _Float16* Wqh = Xv  + n_x;
  _Float16* Wkh = Wqh + n_w;
  _Float16* Wvh = Wkh + n_w;
  _Float16* Woh = Wvh + n_w;
  _Float16* Qh  = Woh + n_w;
  _Float16* Kh  = Qh  + n_x;
  _Float16* Vth = Kh  + n_x;
  _Float16* Ah  = Vth + n_x;

  // fp32 -> fp16 conversions
  cvt_f32_f16<<<2048, 256, 0, stream>>>(q,  Xq,  (int)n_x);
  cvt_f32_f16<<<2048, 256, 0, stream>>>(k,  Xk,  (int)n_x);
  cvt_f32_f16<<<2048, 256, 0, stream>>>(v,  Xv,  (int)n_x);
  cvt_f32_f16<<<1024, 256, 0, stream>>>(Wq, Wqh, (int)n_w);
  cvt_f32_f16<<<1024, 256, 0, stream>>>(Wk, Wkh, (int)n_w);
  cvt_f32_f16<<<1024, 256, 0, stream>>>(Wv, Wvh, (int)n_w);
  cvt_f32_f16<<<1024, 256, 0, stream>>>(Wo, Woh, (int)n_w);

  // projections: 1024 blocks of 8 waves, each block a 128x64 tile
  gemm_wmma<0><<<1024, 256, 0, stream>>>(Xq, Wqh, bq, Qh,  nullptr);
  gemm_wmma<0><<<1024, 256, 0, stream>>>(Xk, Wkh, bk, Kh,  nullptr);
  gemm_wmma<1><<<1024, 256, 0, stream>>>(Xv, Wvh, bv, Vth, nullptr);

  // fused attention: 8192 waves = B*H*(S/16)
  flash_attn<<<1024, 256, 0, stream>>>(Qh, Kh, Vth, Ah);

  // output projection -> fp32 d_out
  gemm_wmma<2><<<1024, 256, 0, stream>>>(Ah, Woh, bo, nullptr, (float*)d_out);
}
